// DeformHash_21397527069370
// MI455X (gfx1250) — matrix operand order repaired and checked by reading, not compiled, not run on hardware
//
#include <hip/hip_runtime.h>

typedef __attribute__((ext_vector_type(16))) _Float16 v16h;
typedef __attribute__((ext_vector_type(8)))  float    v8f;
typedef __attribute__((ext_vector_type(4)))  unsigned int v4u;

#define PRIME1 2654435761u
#define PRIME2 805459861u
#define TBL_MASK 0x7FFFFu      // T-1, T = 1<<19
#define XSTRIDE 72             // padded K-stride in halfs (144B = 36 banks)
#define WFRAG_HALFS (18 * 512) // 9216 halfs = 18432 bytes of weight fragments

union F16Frag { v4u q[2]; v16h v; };

// B fragment: 32 contiguous bytes per lane
__device__ inline v16h load_b_frag(const _Float16* p) {
  F16Frag t;
  t.q[0] = *(const v4u*)(p);
  t.q[1] = *(const v4u*)(p + 8);
  return t.v;
}

// A fragment (16x32 f16): lane<16 -> row=lane, K = kc*32 + {0..7, 16..23}
//                         lane>=16 -> row=lane-16, K = kc*32 + {8..15, 24..31}
__device__ inline v16h load_a_frag(const _Float16* base, int rowBase, int kc, int lane) {
  const _Float16* p = base + (rowBase + (lane & 15)) * XSTRIDE + kc * 32 + ((lane >> 4) << 3);
  F16Frag t;
  t.q[0] = *(const v4u*)(p);
  t.q[1] = *(const v4u*)(p + 16);
  return t.v;
}

// Swizzle W1/W2/W3 (f32, global) into WMMA B-fragment layout (f16) at dst.
// frags 0..7: W1 (kc=frag/4, nt=frag%4, K padded 35->64)
// frags 8..15: W2; frags 16..17: W3 (N padded 2->16)
__device__ inline void swizzle_weights(int start, int stride,
                                       const float* __restrict__ W1,
                                       const float* __restrict__ W2,
                                       const float* __restrict__ W3,
                                       _Float16* dst) {
  for (int i = start; i < WFRAG_HALFS; i += stride) {
    int frag = i >> 9;
    int rem  = i & 511;
    int lane = rem >> 4;
    int s    = rem & 15;
    int kk = ((lane >> 4) << 3) + (s & 7) + ((s >> 3) << 4);   // K within 32-chunk
    int nn = lane & 15;                // N within 16-tile
    float w = 0.f;
    if (frag < 8) {
      int kc = frag >> 2, nt = frag & 3;
      int K = kc * 32 + kk;
      if (K < 35) w = W1[K * 64 + nt * 16 + nn];
    } else if (frag < 16) {
      int f = frag - 8;
      int kc = f >> 2, nt = f & 3;
      w = W2[(kc * 32 + kk) * 64 + nt * 16 + nn];
    } else {
      int kc = frag - 16;
      if (nn < 2) w = W3[(kc * 32 + kk) * 2 + nn];
    }
    dst[(frag << 9) + (lane << 4) + s] = (_Float16)w;
  }
}

// One-shot pre-kernel: build the 18KB weight-fragment image in global scratch.
__global__ __launch_bounds__(256) void weight_swizzle_kernel(
    const float* __restrict__ W1, const float* __restrict__ W2,
    const float* __restrict__ W3, _Float16* __restrict__ wsw) {
  swizzle_weights(blockIdx.x * 256 + threadIdx.x, gridDim.x * 256, W1, W2, W3, wsw);
}

// Encode 8 hash-grid levels [LB, LB+8) for one point into dst (f16 feature row).
template <int LB>
__device__ inline void encodeLevels(float x, float y, float z,
                                    const float* __restrict__ tables,
                                    _Float16* __restrict__ dst) {
  constexpr float kRes[16] = {16.f, 22.f, 30.f, 42.f, 58.f, 80.f, 110.f, 152.f,
                              210.f, 290.f, 400.f, 553.f, 763.f, 1053.f, 1453.f, 2005.f};
#pragma unroll
  for (int li = 0; li < 8; ++li) {
    const int lvl = LB + li;
    const float res = kRes[lvl];
    float px = x * res, py = y * res, pz = z * res;
    float fx = floorf(px), fy = floorf(py), fz = floorf(pz);
    float tx = px - fx, ty = py - fy, tz = pz - fz;
    unsigned cx = (unsigned)(int)fx, cy = (unsigned)(int)fy, cz = (unsigned)(int)fz;
    unsigned hx0 = cx,            hx1 = cx + 1u;
    unsigned hy0 = cy * PRIME1,   hy1 = (cy + 1u) * PRIME1;
    unsigned hz0 = cz * PRIME2,   hz1 = (cz + 1u) * PRIME2;
    float wx0 = 1.f - tx, wy0 = 1.f - ty, wz0 = 1.f - tz;
    const float* tbl = tables + (size_t)lvl * 1048576u;  // T*F floats per level
    float a0 = 0.f, a1 = 0.f;
#pragma unroll
    for (int c = 0; c < 8; ++c) {
      unsigned hx = (c & 4) ? hx1 : hx0;
      unsigned hy = (c & 2) ? hy1 : hy0;
      unsigned hz = (c & 1) ? hz1 : hz0;
      float w = ((c & 4) ? tx : wx0) * ((c & 2) ? ty : wy0) * ((c & 1) ? tz : wz0);
      unsigned idx = (hx ^ hy ^ hz) & TBL_MASK;
      const float2 f = *(const float2*)(tbl + (size_t)idx * 2u);  // L2-resident gather
      a0 += w * f.x;
      a1 += w * f.y;
    }
    dst[3 + 2 * lvl] = (_Float16)a0;
    dst[4 + 2 * lvl] = (_Float16)a1;
  }
}

template <bool PRESWIZZLED>
__global__ __launch_bounds__(256) void deform_hash_mlp_kernel(
    const float* __restrict__ xin, const float* __restrict__ tables,
    const float* __restrict__ W1, const float* __restrict__ W2,
    const float* __restrict__ W3, const _Float16* __restrict__ wsw,
    float* __restrict__ out, int Np) {
  __shared__ alignas(16) _Float16 sX[128 * XSTRIDE];  // encoded inputs, K padded to 64
  __shared__ alignas(16) _Float16 sH[128 * XSTRIDE];  // activations
  __shared__ alignas(16) _Float16 sW[WFRAG_HALFS];    // weight B-fragments
  __shared__ float sOut[8 * 16 * 2];                  // per-wave output staging

  const int tid = threadIdx.x;
  const int blockStart = blockIdx.x * 128;

  // ---- Phase 1a: weight fragments -> LDS ----
  if (PRESWIZZLED) {
    // Branch-free coalesced copy of the pre-swizzled image (1152 x 16B)
    const v4u* src = (const v4u*)wsw;
    v4u* dstv = (v4u*)sW;
#pragma unroll
    for (int i = 0; i < 5; ++i) {
      int idx = tid + i * 256;
      if (idx < WFRAG_HALFS / 8) dstv[idx] = src[idx];
    }
  } else {
    swizzle_weights(tid, 256, W1, W2, W3, sW);
  }

  // ---- Phase 1b: hash-grid encode 128 points, 2 threads per point ----
  {
    int p  = tid >> 1;
    int hp = tid & 1;
    int gp = blockStart + p;
    if (gp >= Np) gp = Np - 1;  // tail: compute garbage, stores are guarded
    float cx = xin[gp * 3 + 0];
    float cy = xin[gp * 3 + 1];
    float cz = xin[gp * 3 + 2];
    _Float16* dst = &sX[p * XSTRIDE];
    if (hp == 0) {
      dst[0] = (_Float16)cx; dst[1] = (_Float16)cy; dst[2] = (_Float16)cz;
      encodeLevels<0>(cx, cy, cz, tables, dst);              // slots 3..18
#pragma unroll
      for (int s = 35; s < 50; ++s) dst[s] = (_Float16)0.f;  // K padding
    } else {
      encodeLevels<8>(cx, cy, cz, tables, dst);              // slots 19..34
#pragma unroll
      for (int s = 50; s < 64; ++s) dst[s] = (_Float16)0.f;
    }
  }
  __syncthreads();

  // ---- Phase 2: per-wave 16-point MLP, v_wmma_f32_16x16x32_f16 ----
  const int wave = tid >> 5;   // wave32
  const int lane = tid & 31;
  const int rowBase = wave << 4;

  v16h a0, a1;

  // Layer 1: [16x64(pad)] @ [64x64] -> relu -> sH
  a0 = load_a_frag(sX, rowBase, 0, lane);
  a1 = load_a_frag(sX, rowBase, 1, lane);
#pragma unroll
  for (int nt = 0; nt < 4; ++nt) {
    v8f acc = {};
    acc = __builtin_amdgcn_wmma_f32_16x16x32_f16(false, a0, false,
            load_b_frag(&sW[((0 * 4 + nt) << 9) + (lane << 4)]), (short)0, acc, false, false);
    acc = __builtin_amdgcn_wmma_f32_16x16x32_f16(false, a1, false,
            load_b_frag(&sW[((1 * 4 + nt) << 9) + (lane << 4)]), (short)0, acc, false, false);
#pragma unroll
    for (int i = 0; i < 8; ++i) {
      float v = fmaxf(acc[i], 0.f);
      sH[(rowBase + ((lane >> 4) << 3) + i) * XSTRIDE + nt * 16 + (lane & 15)] = (_Float16)v;
    }
  }

  // Layer 2: [16x64] @ [64x64] -> relu -> sH (A frags read before overwrite; per-wave
  // LDS ops are in-order, no barrier needed)
  a0 = load_a_frag(sH, rowBase, 0, lane);
  a1 = load_a_frag(sH, rowBase, 1, lane);
#pragma unroll
  for (int nt = 0; nt < 4; ++nt) {
    v8f acc = {};
    acc = __builtin_amdgcn_wmma_f32_16x16x32_f16(false, a0, false,
            load_b_frag(&sW[((8 + nt) << 9) + (lane << 4)]), (short)0, acc, false, false);
    acc = __builtin_amdgcn_wmma_f32_16x16x32_f16(false, a1, false,
            load_b_frag(&sW[((12 + nt) << 9) + (lane << 4)]), (short)0, acc, false, false);
#pragma unroll
    for (int i = 0; i < 8; ++i) {
      float v = fmaxf(acc[i], 0.f);
      sH[(rowBase + ((lane >> 4) << 3) + i) * XSTRIDE + nt * 16 + (lane & 15)] = (_Float16)v;
    }
  }

  // Layer 3: [16x64] @ [64x2(pad16)] -> /5
  a0 = load_a_frag(sH, rowBase, 0, lane);
  a1 = load_a_frag(sH, rowBase, 1, lane);
  {
    v8f acc = {};
    acc = __builtin_amdgcn_wmma_f32_16x16x32_f16(false, a0, false,
            load_b_frag(&sW[(16 << 9) + (lane << 4)]), (short)0, acc, false, false);
    acc = __builtin_amdgcn_wmma_f32_16x16x32_f16(false, a1, false,
            load_b_frag(&sW[(17 << 9) + (lane << 4)]), (short)0, acc, false, false);
    // D layout: lane column n = lane%16 (only n<2 valid), rows m = (lane/16)*8 + i
    if ((lane & 15) < 2) {
      int c = lane & 15;
      int mBase = (lane >> 4) << 3;
#pragma unroll
      for (int i = 0; i < 8; ++i)
        sOut[(rowBase + mBase + i) * 2 + c] = acc[i] * 0.2f;
    }
  }
  __syncthreads();

  // Coalesced f32 store: one lane per output element
  {
    int pt = blockStart + rowBase + (lane >> 1);
    if (pt < Np)
      out[pt * 2 + (lane & 1)] = sOut[(rowBase + (lane >> 1)) * 2 + (lane & 1)];
  }
}

extern "C" void kernel_launch(void* const* d_in, const int* in_sizes, int n_in,
                              void* d_out, int out_size, void* d_ws, size_t ws_size,
                              hipStream_t stream) {
  (void)n_in; (void)out_size;
  const float* x      = (const float*)d_in[0];
  const float* tables = (const float*)d_in[1];
  const float* W1     = (const float*)d_in[2];
  const float* W2     = (const float*)d_in[3];
  const float* W3     = (const float*)d_in[4];
  float* out          = (float*)d_out;
  int Np = in_sizes[0] / 3;
  int blocks = (Np + 127) / 128;

  if (ws_size >= (size_t)(WFRAG_HALFS * sizeof(_Float16))) {
    _Float16* wsw = (_Float16*)d_ws;
    weight_swizzle_kernel<<<9, 256, 0, stream>>>(W1, W2, W3, wsw);
    deform_hash_mlp_kernel<true><<<blocks, 256, 0, stream>>>(
        x, tables, W1, W2, W3, wsw, out, Np);
  } else {
    deform_hash_mlp_kernel<false><<<blocks, 256, 0, stream>>>(
        x, tables, W1, W2, W3, nullptr, out, Np);
  }
}
